// GraphConvolution_46282567582259
// MI455X (gfx1250) — matrix operand order repaired and checked by reading, not compiled, not run on hardware
//
#include <hip/hip_runtime.h>
#include <cstdint>

typedef __bf16 bf16;
typedef __attribute__((ext_vector_type(16))) __bf16 v16bf;
typedef __attribute__((ext_vector_type(8)))  __bf16 v8bf;
typedef __attribute__((ext_vector_type(8)))  float  v8f;
typedef __attribute__((ext_vector_type(4)))  int    v4i;

#define NN 8192
#define SS 8192
#define FF 64
#define OO 64
#define KC 32      // WMMA K per step (bf16)
#define AS 40      // bf16 LDS row stride (80B rows: conflict-free 16B reads)
#define ASF 36     // f32 LDS row stride for 64x32 A tiles (144B rows)
#define BSF 68     // f32 LDS row stride for 32x64 raw inc tiles (272B rows)
#define K0C 0.5f
#define K1C 0.05f
#define K2C (-0.05f)
#define K3C 0.05f

// --------------------------- CDNA5 async copy ------------------------------
// One 16B global->LDS transfer per lane, tracked with ASYNCcnt (ISA §10 async).
typedef __attribute__((address_space(1))) v4i gv4i;   // global int4
typedef __attribute__((address_space(3))) v4i lv4i;   // LDS int4

static __device__ inline void cp_async16(void* lds, const void* g) {
#if __has_builtin(__builtin_amdgcn_global_load_async_to_lds_b128)
  __builtin_amdgcn_global_load_async_to_lds_b128(
      (gv4i*)(uintptr_t)g,
      (lv4i*)(uint32_t)(uintptr_t)lds, 0, 0);
#else
  uint32_t lo = (uint32_t)(uintptr_t)lds;
  asm volatile("global_load_async_to_lds_b128 %0, %1, off"
               :: "v"(lo), "v"(g) : "memory");
#endif
}

#if __has_builtin(__builtin_amdgcn_s_wait_asynccnt)
#define WAIT_ASYNC(n) __builtin_amdgcn_s_wait_asynccnt(n)
#else
#define WAIT_ASYNC(n) asm volatile("s_wait_asynccnt %0" :: "n"(n) : "memory")
#endif

// ------------------------------ WMMA helpers -------------------------------
static __device__ inline v8f wmma_bf16(v16bf a, v16bf b, v8f c) {
  return __builtin_amdgcn_wmma_f32_16x16x32_bf16(false, a, false, b, (short)0, c,
                                                 false, false);
}

// A fragment (16x32 bf16) from bf16 LDS row-major [row][k], stride AS.
static __device__ inline v16bf frag_a(const bf16* A, int m0, int lane) {
  const bf16* p = A + (m0 + (lane & 15)) * AS + ((lane >> 4) << 4);
  v16bf r;
#pragma unroll
  for (int j = 0; j < 8; ++j) r[j] = p[j];
#pragma unroll
  for (int j = 0; j < 8; ++j) r[j + 8] = p[8 + j];
  return r;
}

// A fragment from RAW f32 LDS tile [row][k], stride ASF; cvt to bf16 in regs.
static __device__ inline v16bf frag_a_f32(const float* A, int m0, int lane) {
  const float* p = A + (m0 + (lane & 15)) * ASF + ((lane >> 4) << 4);
  v16bf r;
#pragma unroll
  for (int j = 0; j < 8; ++j) r[j] = (bf16)p[j];
#pragma unroll
  for (int j = 0; j < 8; ++j) r[j + 8] = (bf16)p[8 + j];
  return r;
}

// B fragment (32x16) from bf16 LDS staged K-transposed: Bt[col][k], stride AS.
static __device__ inline v16bf frag_b(const bf16* Bt, int n0, int lane) {
  const bf16* p = Bt + (n0 + (lane & 15)) * AS + ((lane >> 4) << 4);
  v16bf r;
#pragma unroll
  for (int j = 0; j < 16; ++j) r[j] = p[j];
  return r;
}

// B fragment from RAW f32 tile [k][col] (stride BSF): column gather + cvt.
// stride 68 => bank (4k + col) % 64, conflict-free across lanes per k.
static __device__ inline v16bf frag_b_f32col(const float* B, int n0, int lane) {
  const float* p = B + n0 + (lane & 15) + (((lane >> 4) << 4) * BSF);
  v16bf r;
#pragma unroll
  for (int j = 0; j < 16; ++j) r[j] = (bf16)p[j * BSF];
  return r;
}

// ---------------------------------------------------------------------------
// K1: xT[f][n] = bf16(x[n][f])
// ---------------------------------------------------------------------------
__global__ __launch_bounds__(256) void prep_xT(const float* __restrict__ x,
                                               bf16* __restrict__ xT) {
  int n = blockIdx.x * 256 + threadIdx.x;
  int f = blockIdx.y;
  xT[(size_t)f * NN + n] = (bf16)x[(size_t)n * FF + f];
}

// ---------------------------------------------------------------------------
// K2: smT = x^T @ inc (WMMA), fused per-column size/first metadata.
// inc tiles stream via async-to-LDS, double-buffered, raw fp32 in LDS.
// ---------------------------------------------------------------------------
__global__ __launch_bounds__(256) void gemm_inc_t(const float* __restrict__ inc,
                                                  const bf16* __restrict__ xT,
                                                  float* __restrict__ smT,
                                                  float* __restrict__ sizef,
                                                  int* __restrict__ firstI) {
  __shared__ alignas(16) bf16  Axt[2][64 * AS];   // xT chunk (bf16, ready)
  __shared__ alignas(16) float Bi[2][32 * BSF];   // inc chunk RAW [k][s]
  const int t = threadIdx.x, lane = t & 31, wave = t >> 5;
  const int mt = wave & 3, np = wave >> 2;
  const int s0 = blockIdx.x * 64;
  const int sc = t & 63, kb = (t >> 6) * 8;       // metadata role
  v8f a0 = {}, a1 = {};
  int cnt = 0, first = -1;

  auto issue = [&](int k, int b) {
    cp_async16(&Axt[b][(t >> 2) * AS + (t & 3) * 8],
               xT + (size_t)(t >> 2) * NN + k + (t & 3) * 8);
#pragma unroll
    for (int i = 0; i < 2; ++i) {
      int id = t + 256 * i;
      int r = id >> 4, c = id & 15;               // 32 rows x 16 chunks
      cp_async16(&Bi[b][r * BSF + c * 4],
                 inc + (size_t)(k + r) * SS + s0 + c * 4);
    }
  };

  issue(0, 0);
  int buf = 0;
  for (int k = 0; k < NN; k += KC) {
    if (k + KC < NN) { issue(k + KC, buf ^ 1); WAIT_ASYNC(3); }
    else             { WAIT_ASYNC(0); }
    __syncthreads();
    // metadata scan on the raw tile (4 threads per column)
#pragma unroll
    for (int j = 0; j < 8; ++j) {
      float v = Bi[buf][(kb + j) * BSF + sc];
      cnt += (v != 0.0f) ? 1 : 0;
      if (first < 0 && v != 0.0f) first = k + kb + j;
    }
    v16bf fa = frag_a(Axt[buf], mt * 16, lane);
    v16bf b0 = frag_b_f32col(Bi[buf], np * 32, lane);
    v16bf b1 = frag_b_f32col(Bi[buf], np * 32 + 16, lane);
    a0 = wmma_bf16(fa, b0, a0);
    a1 = wmma_bf16(fa, b1, a1);
    __syncthreads();
    buf ^= 1;
  }
#pragma unroll
  for (int r = 0; r < 8; ++r) {
    int f = mt * 16 + ((lane >> 4) << 3) + r;
    int c0 = s0 + np * 32 + (lane & 15);
    smT[(size_t)f * SS + c0] = a0[r];
    smT[(size_t)f * SS + c0 + 16] = a1[r];
  }
  // reduce metadata (4 kb-groups per column) through LDS
  int* mcnt = (int*)&Axt[0][0];
  int* mfst = (int*)&Bi[0][0];
  mcnt[t] = cnt;
  mfst[t] = first;
  __syncthreads();
  if (t < 64) {
    int c = 0, f = 0x7fffffff;
#pragma unroll
    for (int g = 0; g < 4; ++g) {
      c += mcnt[t + 64 * g];
      int ff = mfst[t + 64 * g];
      if (ff >= 0 && ff < f) f = ff;
    }
    sizef[s0 + t] = (float)c;
    firstI[s0 + t] = (f == 0x7fffffff) ? 0 : f;
  }
}

// ---------------------------------------------------------------------------
// K3: out_pre = K0*x + K1*sin(lapP@x) + K2*sin(lapN@x).
// Both 256 MB Laplacians stream once via async-to-LDS, double-buffered.
// ---------------------------------------------------------------------------
__global__ __launch_bounds__(256) void gemm_lap(const float* __restrict__ x,
                                                const float* __restrict__ lapP,
                                                const float* __restrict__ lapN,
                                                const bf16* __restrict__ xT,
                                                float* __restrict__ outPre) {
  __shared__ alignas(16) float Ap[2][64 * ASF];
  __shared__ alignas(16) float An[2][64 * ASF];
  __shared__ alignas(16) bf16  Bt[2][64 * AS];
  const int t = threadIdx.x, lane = t & 31, wave = t >> 5;
  const int mt = wave & 3, np = wave >> 2;
  const int n0 = blockIdx.x * 64;
  v8f aP0 = {}, aP1 = {}, aN0 = {}, aN1 = {};

  auto issue = [&](int k, int b) {
#pragma unroll
    for (int i = 0; i < 2; ++i) {
      int id = t + 256 * i;
      int r = id >> 3, c = id & 7;                // 64 rows x 8 chunks
      cp_async16(&Ap[b][r * ASF + c * 4], lapP + (size_t)(n0 + r) * NN + k + c * 4);
      cp_async16(&An[b][r * ASF + c * 4], lapN + (size_t)(n0 + r) * NN + k + c * 4);
    }
    cp_async16(&Bt[b][(t >> 2) * AS + (t & 3) * 8],
               xT + (size_t)(t >> 2) * NN + k + (t & 3) * 8);
  };

  issue(0, 0);
  int buf = 0;
  for (int k = 0; k < NN; k += KC) {
    if (k + KC < NN) { issue(k + KC, buf ^ 1); WAIT_ASYNC(5); }
    else             { WAIT_ASYNC(0); }
    __syncthreads();
    v16bf b0 = frag_b(Bt[buf], np * 32, lane);
    v16bf b1 = frag_b(Bt[buf], np * 32 + 16, lane);
    v16bf fp = frag_a_f32(Ap[buf], mt * 16, lane);
    v16bf fn = frag_a_f32(An[buf], mt * 16, lane);
    aP0 = wmma_bf16(fp, b0, aP0);
    aP1 = wmma_bf16(fp, b1, aP1);
    aN0 = wmma_bf16(fn, b0, aN0);
    aN1 = wmma_bf16(fn, b1, aN1);
    __syncthreads();
    buf ^= 1;
  }
#pragma unroll
  for (int r = 0; r < 8; ++r) {
    int m = n0 + mt * 16 + ((lane >> 4) << 3) + r;
    int f0 = np * 32 + (lane & 15);
    size_t o0 = (size_t)m * FF + f0;
    size_t o1 = o0 + 16;
    outPre[o0] = K0C * x[o0] + K1C * sinf(aP0[r]) + K2C * sinf(aN0[r]);
    outPre[o1] = K0C * x[o1] + K1C * sinf(aP1[r]) + K2C * sinf(aN1[r]);
  }
}

// ---------------------------------------------------------------------------
// K4: ttT[f][s] = bf16( valid(size) * tanh(size*x[first][f] - smT[f][s]) )
// ---------------------------------------------------------------------------
__global__ __launch_bounds__(256) void build_tt(const float* __restrict__ x,
                                                const float* __restrict__ smT,
                                                const float* __restrict__ sizef,
                                                const int* __restrict__ firstI,
                                                bf16* __restrict__ ttT) {
  int s = blockIdx.x * 256 + threadIdx.x;
  int f = blockIdx.y;
  float sz = sizef[s];
  int fi = firstI[s];
  float tv = tanhf(sz * x[(size_t)fi * FF + f] - smT[(size_t)f * SS + s]);
  bool valid = (sz == 3.0f) | (sz == 4.0f) | (sz == 10.0f);
  ttT[(size_t)f * SS + s] = (bf16)(valid ? tv : 0.0f);
}

// ---------------------------------------------------------------------------
// K5: acc = inc @ tt (async double-buffered), fused epilogue:
//     P = out_pre + K3*acc  ->  out = P @ weight + bias  (WMMA in-tile)
// ---------------------------------------------------------------------------
__global__ __launch_bounds__(256) void gemm_d(const float* __restrict__ inc,
                                              const bf16* __restrict__ ttT,
                                              const float* __restrict__ outPre,
                                              const float* __restrict__ weight,
                                              const float* __restrict__ bias,
                                              float* __restrict__ out) {
  __shared__ alignas(16) float Ai[2][64 * ASF];
  __shared__ alignas(16) bf16  Bt[2][64 * AS];
  __shared__ alignas(16) bf16  Wt[64 * AS];   // Wt[o][f] = weight[f][o]
  __shared__ alignas(16) bf16  Pbf[64 * AS];  // pre-activation tile, bf16
  const int t = threadIdx.x, lane = t & 31, wave = t >> 5;
  const int mt = wave & 3, np = wave >> 2;
  const int n0 = blockIdx.x * 64;
  {
    int o = t >> 2, p4 = t & 3;
    v8bf w;
#pragma unroll
    for (int j = 0; j < 8; ++j) w[j] = (bf16)weight[(size_t)(p4 * 8 + j) * OO + o];
    *(v8bf*)&Wt[o * AS + p4 * 8] = w;
  }
  v8f a0 = {}, a1 = {};

  auto issue = [&](int k, int b) {
#pragma unroll
    for (int i = 0; i < 2; ++i) {
      int id = t + 256 * i;
      int r = id >> 3, c = id & 7;
      cp_async16(&Ai[b][r * ASF + c * 4], inc + (size_t)(n0 + r) * SS + k + c * 4);
    }
    cp_async16(&Bt[b][(t >> 2) * AS + (t & 3) * 8],
               ttT + (size_t)(t >> 2) * SS + k + (t & 3) * 8);
  };

  issue(0, 0);
  int buf = 0;
  for (int k = 0; k < SS; k += KC) {
    if (k + KC < SS) { issue(k + KC, buf ^ 1); WAIT_ASYNC(3); }
    else             { WAIT_ASYNC(0); }
    __syncthreads();
    v16bf fa = frag_a_f32(Ai[buf], mt * 16, lane);
    v16bf b0 = frag_b(Bt[buf], np * 32, lane);
    v16bf b1 = frag_b(Bt[buf], np * 32 + 16, lane);
    a0 = wmma_bf16(fa, b0, a0);
    a1 = wmma_bf16(fa, b1, a1);
    __syncthreads();
    buf ^= 1;
  }
  // epilogue: P = out_pre + K3*acc -> bf16 tile
#pragma unroll
  for (int r = 0; r < 8; ++r) {
    int ml = mt * 16 + ((lane >> 4) << 3) + r;
    int f0 = np * 32 + (lane & 15);
    float pv0 = outPre[(size_t)(n0 + ml) * FF + f0] + K3C * a0[r];
    float pv1 = outPre[(size_t)(n0 + ml) * FF + f0 + 16] + K3C * a1[r];
    Pbf[ml * AS + f0] = (bf16)pv0;
    Pbf[ml * AS + f0 + 16] = (bf16)pv1;
  }
  __syncthreads();
  // projection: out = P @ weight + bias
  v8f w0 = {}, w1 = {};
#pragma unroll
  for (int kf = 0; kf < FF; kf += KC) {
    v16bf fa = frag_a(Pbf + kf, mt * 16, lane);
    v16bf b0 = frag_b(Wt + kf, np * 32, lane);
    v16bf b1 = frag_b(Wt + kf, np * 32 + 16, lane);
    w0 = wmma_bf16(fa, b0, w0);
    w1 = wmma_bf16(fa, b1, w1);
  }
#pragma unroll
  for (int r = 0; r < 8; ++r) {
    int m = n0 + mt * 16 + ((lane >> 4) << 3) + r;
    int o0 = np * 32 + (lane & 15);
    out[(size_t)m * OO + o0] = w0[r] + bias[o0];
    out[(size_t)m * OO + o0 + 16] = w1[r] + bias[o0 + 16];
  }
}

// ---------------------------------------------------------------------------
extern "C" void kernel_launch(void* const* d_in, const int* in_sizes, int n_in,
                              void* d_out, int out_size, void* d_ws, size_t ws_size,
                              hipStream_t stream) {
  (void)in_sizes; (void)n_in; (void)out_size; (void)ws_size;
  const float* x    = (const float*)d_in[0];
  const float* lapP = (const float*)d_in[1];
  const float* lapN = (const float*)d_in[2];
  const float* inc  = (const float*)d_in[3];
  const float* wgt  = (const float*)d_in[4];
  const float* bias = (const float*)d_in[5];
  float* out = (float*)d_out;

  char* ws = (char*)d_ws;
  bf16*  xT     = (bf16*)(ws);                                  // 1 MB
  float* smT    = (float*)(ws + (size_t)(1u << 20));            // 2 MB
  bf16*  ttT    = (bf16*)(ws + (size_t)3 * (1u << 20));         // 1 MB
  float* outPre = (float*)(ws + (size_t)4 * (1u << 20));        // 2 MB
  float* sizef  = (float*)(ws + (size_t)6 * (1u << 20));        // 32 KB
  int*   firstI = (int*)(ws + (size_t)6 * (1u << 20) + 32768);  // 32 KB

  prep_xT   <<<dim3(NN / 256, FF), 256, 0, stream>>>(x, xT);
  gemm_inc_t<<<dim3(SS / 64), 256, 0, stream>>>(inc, xT, smT, sizef, firstI);
  gemm_lap  <<<dim3(NN / 64), 256, 0, stream>>>(x, lapP, lapN, xT, outPre);
  build_tt  <<<dim3(SS / 256, FF), 256, 0, stream>>>(x, smT, sizef, firstI, ttT);
  gemm_d    <<<dim3(NN / 64), 256, 0, stream>>>(inc, ttT, outPre, wgt, bias, out);
}
